// GTV_7610682048796
// MI455X (gfx1250) — matrix operand order repaired and checked by reading, not compiled, not run on hardware
//
#include <hip/hip_runtime.h>
#include <hip/hip_bf16.h>

typedef __attribute__((ext_vector_type(16))) _Float16 v16h;
typedef __attribute__((ext_vector_type(8)))  _Float16 v8h;
typedef __attribute__((ext_vector_type(8)))  float    v8f;

#define WIDTH 36
#define NN    1296      // nodes per image
#define EE    2520      // edges per image
#define BB    4
#define EH    1260      // horizontal edges (y*35+x); vertical edges start at 1260
#define ACT_STRIDE 32   // channel-last activation stride (zero-padded)

// ---------------------------------------------------------------------------
// Pack xf [B,3,36,36] f32 -> channel-last f16 act [B][1296][32], ch 3..31 = 0
// ---------------------------------------------------------------------------
__global__ void pack_input(const float* __restrict__ xf, _Float16* __restrict__ act)
{
    int idx = blockIdx.x * 256 + threadIdx.x;       // over B*NN
    if (idx >= BB * NN) return;
    int b = idx / NN, p = idx % NN;
    _Float16* dst = act + idx * ACT_STRIDE;
#pragma unroll
    for (int ch = 0; ch < 32; ++ch)
        dst[ch] = (ch < 3) ? (_Float16)xf[(b * 3 + ch) * NN + p] : (_Float16)0.0f;
}

// ---------------------------------------------------------------------------
// Conv3x3 SAME as implicit GEMM on WMMA f16->f32.
// Block = 96 threads = 3 waves = 48 consecutive pixels (48 | 1296: never
// crosses batch; 48 = 36+12: spans exactly 2 image rows). Grid = 108 blocks.
// LDS: weights packed in B-fragment layout [tap][co][32ch], input staged as a
// zero-haloed [4][38][32] f16 tile. Inner loop: pure ds_load_b128 + v_wmma,
// no divergence (EXEC all ones, as WMMA requires).
// ---------------------------------------------------------------------------
template<int CIN, int COUT, bool RELU>
__global__ __launch_bounds__(96) void conv3x3_wmma(
    const _Float16* __restrict__ in,    // [B][1296][32] channel-last f16
    const float*    __restrict__ wt,    // [COUT][CIN][3][3] f32
    const float*    __restrict__ bias,  // [COUT]
    _Float16*       __restrict__ out)   // [B][1296][32] channel-last f16
{
    __shared__ __align__(32) _Float16 wl[9 * 32 * 32];   // 18 KB
    __shared__ __align__(32) _Float16 tile[4 * 38 * 32]; // 9.5 KB
    const int tid = threadIdx.x;

    // ---- pack weights into WMMA B layout: wl[tap][co][kc] ----
    for (int i = tid; i < 9 * 32 * 32; i += 96) {
        int tap = i >> 10, rem = i & 1023, co = rem >> 5, kc = rem & 31;
        float v = (co < COUT && kc < CIN) ? wt[(co * CIN + kc) * 9 + tap] : 0.0f;
        wl[i] = (_Float16)v;
    }

    // ---- stage zero-haloed activation tile ----
    const int pb_blk = blockIdx.x * 48;
    const int b      = pb_blk / NN;
    const int p0     = pb_blk % NN;
    const int y0     = p0 / WIDTH;           // block rows: y0, y0+1
    const _Float16* actb = in + b * NN * ACT_STRIDE;
    for (int i = tid; i < 4 * 38; i += 96) {
        int rr = i / 38, cc = i % 38;
        int gy = y0 + rr - 1, gx = cc - 1;
        _Float16* dst = tile + i * 32;
        if (gy >= 0 && gy < WIDTH && gx >= 0 && gx < WIDTH) {
            const _Float16* src = actb + (gy * WIDTH + gx) * ACT_STRIDE;
            *(uint4*)(dst)      = *(const uint4*)(src);
            *(uint4*)(dst + 8)  = *(const uint4*)(src + 8);
            *(uint4*)(dst + 16) = *(const uint4*)(src + 16);
            *(uint4*)(dst + 24) = *(const uint4*)(src + 24);
        } else {
            uint4 z = {0u, 0u, 0u, 0u};
            *(uint4*)(dst)      = z;
            *(uint4*)(dst + 8)  = z;
            *(uint4*)(dst + 16) = z;
            *(uint4*)(dst + 24) = z;
        }
    }
    __syncthreads();

    const int  wave = tid >> 5;
    const int  lane = tid & 31;
    const bool hi   = lane >= 16;
    const int  n    = lane & 15;                 // A row within tile / B,D column
    const int  p    = p0 + wave * 16 + n;        // this lane's pixel (A row)
    const int  rb   = p / WIDTH - y0;            // 0 or 1
    const int  pcol = p % WIDTH;
    const int  c0   = hi ? 16 : 0;               // A-layout channel base (contiguous k = 16h + s, same order as B)
    const int  klo  = hi ? 16 : 0;               // B-layout K base

    v8f acc0 = {}, acc1 = {};
#pragma unroll
    for (int t = 0; t < 9; ++t) {
        const int dy = t / 3, dx = t % 3;
        const int ai = ((rb + dy) * 38 + (pcol + dx)) * 32 + c0;
        v8h  alo = *(const v8h*)&tile[ai];
        v8h  ahi = *(const v8h*)&tile[ai + 8];
        v16h a   = __builtin_shufflevector(alo, ahi,
                       0, 1, 2, 3, 4, 5, 6, 7, 8, 9, 10, 11, 12, 13, 14, 15);
        v16h b0  = *(const v16h*)&wl[(t << 10) + (n << 5) + klo];
        acc0 = __builtin_amdgcn_wmma_f32_16x16x32_f16(false, a, false, b0,
                                                      (short)0, acc0, false, false);
        if constexpr (COUT > 16) {
            v16h b1 = *(const v16h*)&wl[(t << 10) + ((16 + n) << 5) + klo];
            acc1 = __builtin_amdgcn_wmma_f32_16x16x32_f16(false, a, false, b1,
                                                          (short)0, acc1, false, false);
        }
    }

    // ---- store channel-last f16 (D layout: vgpr r -> M = r + (hi?8:0)) ----
    const int opix0 = p0 + wave * 16;
#pragma unroll
    for (int r = 0; r < 8; ++r) {
        const int mm = r + (hi ? 8 : 0);
        _Float16* orow = out + (b * NN + opix0 + mm) * ACT_STRIDE;
        if (n < COUT) {
            float v = acc0[r] + bias[n];
            if (RELU) v = fmaxf(v, 0.0f);
            orow[n] = (_Float16)v;
        }
        if constexpr (COUT > 16) {
            float v = acc1[r] + bias[16 + n];
            if (RELU) v = fmaxf(v, 0.0f);
            orow[16 + n] = (_Float16)v;
        }
    }
}

// ---------------------------------------------------------------------------
// Edge weights: wg[b,e] = exp(-sum_k (f[b,k,ci]-f[b,k,cj])^2 / sigma^2)
// e<1260 -> horizontal (y*35+x): i=y*36+x, j=i+1 ; else vertical: i=e-1260, j=i+36
// ---------------------------------------------------------------------------
__device__ __forceinline__ void edge_nodes(int e, int& i, int& j)
{
    if (e < EH) { int y = e / 35, x = e % 35; i = y * WIDTH + x; j = i + 1; }
    else        { i = e - EH;                                    j = i + WIDTH; }
}

__global__ void edge_weights(const _Float16* __restrict__ feat,  // [B][1296][32], ch<6 valid
                             float* __restrict__ wg)             // [B][2520]
{
    int idx = blockIdx.x * blockDim.x + threadIdx.x;
    if (idx >= BB * EE) return;
    int b = idx / EE, e = idx % EE;
    int i, j; edge_nodes(e, i, j);
    const _Float16* fi = feat + (b * NN + i) * ACT_STRIDE;
    const _Float16* fj = feat + (b * NN + j) * ACT_STRIDE;
    float s = 0.0f;
#pragma unroll
    for (int k = 0; k < 6; ++k) {
        float d = (float)fi[k] - (float)fj[k];
        s += d * d;
    }
    wg[idx] = __expf(-s * 1.0e4f);   // / sigma^2 = / 1e-4
}

// ---------------------------------------------------------------------------
// 9 x solve_step. A = I + u*L (weighted 5-point grid Laplacian, SPD) solved by
// Jacobi-preconditioned CG entirely in LDS. One workgroup per (b,c) system.
// ---------------------------------------------------------------------------
#define CG_ITERS 64

__device__ __forceinline__ float lap_apply(const float* __restrict__ v,
                                           const float* __restrict__ g,
                                           int i, float u)
{
    int y = i / WIDTH, x = i % WIDTH;
    float vi  = v[i];
    float acc = 0.0f;
    if (x < WIDTH - 1) acc += g[y * 35 + x]         * (vi - v[i + 1]);
    if (x > 0)         acc += g[y * 35 + x - 1]     * (vi - v[i - 1]);
    if (y < WIDTH - 1) acc += g[EH + i]             * (vi - v[i + WIDTH]);
    if (y > 0)         acc += g[EH + i - WIDTH]     * (vi - v[i - WIDTH]);
    return vi + u * acc;
}

__device__ __forceinline__ float node_deg(const float* __restrict__ g, int i)
{
    int y = i / WIDTH, x = i % WIDTH;
    float d = 0.0f;
    if (x < WIDTH - 1) d += g[y * 35 + x];
    if (x > 0)         d += g[y * 35 + x - 1];
    if (y < WIDTH - 1) d += g[EH + i];
    if (y > 0)         d += g[EH + i - WIDTH];
    return d;
}

__device__ __forceinline__ float block_reduce(float s, float* red, int tid)
{
    red[tid] = s;
    __syncthreads();
    for (int off = 128; off > 0; off >>= 1) {
        if (tid < off) red[tid] += red[tid + off];
        __syncthreads();
    }
    float r = red[0];
    __syncthreads();
    return r;
}

__global__ __launch_bounds__(256) void glr_solve(
    const float* __restrict__ xf,   // [B,3,1296]
    const float* __restrict__ wg,   // [B,2520]
    const float* __restrict__ up,   // [1]
    float* __restrict__ out)        // [B,3,1296]
{
    __shared__ float xv[NN], sol[NN], rv[NN], pv[NN], apv[NN], dinv[NN];
    __shared__ float ge[EE];
    __shared__ float red[256];

    const int tid = threadIdx.x;
    const int bc  = blockIdx.x;         // b*3 + c
    const int b   = bc / 3;
    const float* y0  = xf + bc * NN;
    const float* wgb = wg + b * EE;
    float u = up[0];
    u = fminf(fmaxf(u, 0.001f), 1.0f);

    for (int i = tid; i < NN; i += 256) xv[i] = y0[i];
    __syncthreads();

    for (int outer = 0; outer < 9; ++outer) {
        for (int e = tid; e < EE; e += 256) {
            int i, j; edge_nodes(e, i, j);
            float z = fmaxf(fabsf(xv[i] - xv[j]), 0.01f);
            ge[e] = wgb[e] / z;
        }
        __syncthreads();

        for (int i = tid; i < NN; i += 256) {
            dinv[i] = 1.0f / (1.0f + u * node_deg(ge, i));
            sol[i]  = xv[i];
        }
        __syncthreads();

        float rz_part = 0.0f;
        for (int i = tid; i < NN; i += 256) {
            float r = xv[i] - lap_apply(xv, ge, i, u);
            rv[i] = r;
            float z = dinv[i] * r;
            pv[i] = z;
            rz_part += r * z;
        }
        __syncthreads();
        float rz = block_reduce(rz_part, red, tid);

        for (int it = 0; it < CG_ITERS; ++it) {
            for (int i = tid; i < NN; i += 256) apv[i] = lap_apply(pv, ge, i, u);
            __syncthreads();

            float pap_part = 0.0f;
            for (int i = tid; i < NN; i += 256) pap_part += pv[i] * apv[i];
            float pap = block_reduce(pap_part, red, tid);

            float alpha = rz / (pap + 1.0e-30f);

            float rz_new_part = 0.0f;
            for (int i = tid; i < NN; i += 256) {
                sol[i] += alpha * pv[i];
                float r = rv[i] - alpha * apv[i];
                rv[i] = r;
                rz_new_part += r * dinv[i] * r;
            }
            float rz_new = block_reduce(rz_new_part, red, tid);

            float beta = rz_new / (rz + 1.0e-30f);
            rz = rz_new;

            for (int i = tid; i < NN; i += 256)
                pv[i] = dinv[i] * rv[i] + beta * pv[i];
            __syncthreads();
        }

        for (int i = tid; i < NN; i += 256) xv[i] = sol[i];
        __syncthreads();
    }

    float* o = out + bc * NN;
    for (int i = tid; i < NN; i += 256) o[i] = xv[i];
}

// ---------------------------------------------------------------------------
// Host-side launcher
// ---------------------------------------------------------------------------
extern "C" void kernel_launch(void* const* d_in, const int* in_sizes, int n_in,
                              void* d_out, int out_size, void* d_ws, size_t ws_size,
                              hipStream_t stream)
{
    // setup_inputs order: xf, H, ci, cj, u_param, then (w0,b0)...(w6,b6)
    const float* xf = (const float*)d_in[0];
    const float* up = (const float*)d_in[4];
    const float* w[7];
    const float* bi[7];
    for (int i = 0; i < 7; ++i) {
        w[i]  = (const float*)d_in[5 + 2 * i];
        bi[i] = (const float*)d_in[6 + 2 * i];
    }

    const size_t ACT_ELEMS = (size_t)BB * NN * ACT_STRIDE;   // 165888 halves
    _Float16* actA  = (_Float16*)d_ws;
    _Float16* actB  = actA + ACT_ELEMS;
    float*    wgbuf = (float*)(actB + ACT_ELEMS);

    pack_input<<<(BB * NN + 255) / 256, 256, 0, stream>>>(xf, actB);

    // 7 conv layers, ping-pong. Grid: 108 blocks x 96 threads = 324 wave-tiles.
    conv3x3_wmma<3, 32, true  ><<<108, 96, 0, stream>>>(actB, w[0], bi[0], actA);
    conv3x3_wmma<32, 32, true ><<<108, 96, 0, stream>>>(actA, w[1], bi[1], actB);
    conv3x3_wmma<32, 32, true ><<<108, 96, 0, stream>>>(actB, w[2], bi[2], actA);
    conv3x3_wmma<32, 32, true ><<<108, 96, 0, stream>>>(actA, w[3], bi[3], actB);
    conv3x3_wmma<32, 32, true ><<<108, 96, 0, stream>>>(actB, w[4], bi[4], actA);
    conv3x3_wmma<32, 32, true ><<<108, 96, 0, stream>>>(actA, w[5], bi[5], actB);
    conv3x3_wmma<32, 6,  false><<<108, 96, 0, stream>>>(actB, w[6], bi[6], actA);

    edge_weights<<<(BB * EE + 255) / 256, 256, 0, stream>>>(actA, wgbuf);

    glr_solve<<<BB * 3, 256, 0, stream>>>(xf, wgbuf, up, (float*)d_out);
}
